// NnBoard768_45835890983582
// MI455X (gfx1250) — matrix operand order repaired and checked by reading, not compiled, not run on hardware
//
#include <hip/hip_runtime.h>
#include <math.h>

typedef __attribute__((ext_vector_type(16))) _Float16 v16h;
typedef __attribute__((ext_vector_type(8)))  float    v8f;

#define FT_OUT  512
#define NFEAT   768
#define MROWS   32            // batch rows per WG (two 16-row WMMA M-tiles)
#define ASTRIDE (NFEAT + 4)   // 772 floats: rows land 4 banks apart -> conflict-free
#define KTILES  (NFEAT / 32)  // 24
#define NTILES_PER_WAVE 4     // 8 waves * 4 tiles * 16 = 512 outputs

// ---------------------------------------------------------------------------
// Pre-pass: ft_w [K=768][N=512] fp32  ->  w16 [N=512][K=768] fp16 (transposed)
// so each lane's WMMA B-fragment is K-contiguous (one aligned 32B load).
// ---------------------------------------------------------------------------
__global__ __launch_bounds__(256) void nn_convert_w(const float* __restrict__ ft_w,
                                                    _Float16* __restrict__ w16) {
    int i = blockIdx.x * 256 + threadIdx.x;
    if (i < FT_OUT * NFEAT) {
        int n = i / NFEAT;
        int k = i - n * NFEAT;
        w16[i] = (_Float16)ft_w[(size_t)k * FT_OUT + n];
    }
}

// ---------------------------------------------------------------------------
// Fused NNUE forward: densify 32 batch rows of sparse features into LDS,
// WMMA against fp16 weights (each B fragment reused by 2 M-tiles),
// clip+dot+sigmoid, write 32 floats per WG.
// ---------------------------------------------------------------------------
__global__ __launch_bounds__(256) void nn_forward(
    const int*      __restrict__ stm_idx,   // [2, nnz] flat: bids | feats
    const int*      __restrict__ nstm_idx,  // [2, nnz]
    const float*    __restrict__ values,    // [nnz]
    const float*    __restrict__ ft_b,      // [512]
    const float*    __restrict__ out_w,     // [1024]
    const float*    __restrict__ out_b,     // [1]
    const _Float16* __restrict__ w16,       // [512][768] fp16
    float*          __restrict__ out,       // [B]
    int nnz, int mp, int batch)
{
    __shared__ float Alds[MROWS * ASTRIDE];   // ~98.8 KB (CDNA5 WGP: 320 KB)
    __shared__ float dotAcc[MROWS];

    const int tid   = threadIdx.x;
    const int wg    = blockIdx.x;
    const int lane  = tid & 31;
    const int wave  = tid >> 5;        // 0..7
    const int hi    = lane >> 4;       // 0: lanes 0-15, 1: lanes 16-31
    const int mlane = lane & 15;       // N column within tile / M row for A

    if (tid < MROWS) dotAcc[tid] = 0.0f;

    const int ebase = wg * MROWS * mp;           // first nnz entry for this WG
    const int ecnt  = MROWS * mp;

    #pragma unroll
    for (int pass = 0; pass < 2; ++pass) {
        const int* bids  = (pass == 0) ? stm_idx  : nstm_idx;
        const int* feats = bids + nnz;

        __syncthreads();   // previous pass done reading A before we clobber it
        for (int i = tid; i < MROWS * ASTRIDE; i += 256) Alds[i] = 0.0f;
        __syncthreads();

        // scatter: A[m][feat] += value  (duplicates accumulate as multiplicity)
        for (int e = tid; e < ecnt; e += 256) {
            int g = ebase + e;
            if (g < nnz) {
                int f = feats[g];
                int m = bids[g] - wg * MROWS;
                if (m >= 0 && m < MROWS && f >= 0 && f < NFEAT)
                    atomicAdd(&Alds[m * ASTRIDE + f], values[g]);
            }
        }
        __syncthreads();

        // --- WMMA: A(32x768) x W(768x512); wave owns 4 N-tiles x 2 M-tiles ---
        v8f c00 = {}, c01 = {}, c02 = {}, c03 = {};   // M-tile 0 (rows 0..15)
        v8f c10 = {}, c11 = {}, c12 = {}, c13 = {};   // M-tile 1 (rows 16..31)
        const int nb = wave * NTILES_PER_WAVE * 16;   // base N for this wave

        for (int kt = 0; kt < KTILES; ++kt) {
            const int koff = kt * 32;
            // A fragment: lane holds row M=mlane; element j -> K = koff + hi*16
            //   + j (contiguous, matches B)
            const float* arow0 = &Alds[mlane        * ASTRIDE + koff + hi * 16];
            const float* arow1 = &Alds[(mlane + 16) * ASTRIDE + koff + hi * 16];
            v16h a0, a1;
            #pragma unroll
            for (int j = 0; j < 8; ++j) {
                a0[j]     = (_Float16)arow0[j];
                a0[j + 8] = (_Float16)arow0[8 + j];
                a1[j]     = (_Float16)arow1[j];
                a1[j + 8] = (_Float16)arow1[8 + j];
            }
            // B fragments: lane holds column N; element j -> K = koff + hi*16 + j
            const _Float16* wbase = w16 + (size_t)koff + (size_t)hi * 16;
            const v16h b0v = *(const v16h*)(wbase + (size_t)(nb +  0 + mlane) * NFEAT);
            const v16h b1v = *(const v16h*)(wbase + (size_t)(nb + 16 + mlane) * NFEAT);
            const v16h b2v = *(const v16h*)(wbase + (size_t)(nb + 32 + mlane) * NFEAT);
            const v16h b3v = *(const v16h*)(wbase + (size_t)(nb + 48 + mlane) * NFEAT);
            // each B fragment feeds both M-tiles (2x arithmetic intensity on B)
            c00 = __builtin_amdgcn_wmma_f32_16x16x32_f16(false, a0, false, b0v, (short)0, c00, false, false);
            c10 = __builtin_amdgcn_wmma_f32_16x16x32_f16(false, a1, false, b0v, (short)0, c10, false, false);
            c01 = __builtin_amdgcn_wmma_f32_16x16x32_f16(false, a0, false, b1v, (short)0, c01, false, false);
            c11 = __builtin_amdgcn_wmma_f32_16x16x32_f16(false, a1, false, b1v, (short)0, c11, false, false);
            c02 = __builtin_amdgcn_wmma_f32_16x16x32_f16(false, a0, false, b2v, (short)0, c02, false, false);
            c12 = __builtin_amdgcn_wmma_f32_16x16x32_f16(false, a1, false, b2v, (short)0, c12, false, false);
            c03 = __builtin_amdgcn_wmma_f32_16x16x32_f16(false, a0, false, b3v, (short)0, c03, false, false);
            c13 = __builtin_amdgcn_wmma_f32_16x16x32_f16(false, a1, false, b3v, (short)0, c13, false, false);
        }

        // --- bias + clip[0,1] + partial dot with out_w ---
        // C layout: element r -> M = r + hi*8 (+16 for M-tile 1), N = base + mlane
        float dp0[8] = {0.f,0.f,0.f,0.f,0.f,0.f,0.f,0.f};
        float dp1[8] = {0.f,0.f,0.f,0.f,0.f,0.f,0.f,0.f};
        const v8f* cs0[4] = { &c00, &c01, &c02, &c03 };
        const v8f* cs1[4] = { &c10, &c11, &c12, &c13 };
        #pragma unroll
        for (int q = 0; q < NTILES_PER_WAVE; ++q) {
            const int n  = nb + q * 16 + mlane;
            const float bn = ft_b[n];
            const float wn = out_w[pass * FT_OUT + n];
            #pragma unroll
            for (int r = 0; r < 8; ++r) {
                float h0 = (*cs0[q])[r] + bn;
                float h1 = (*cs1[q])[r] + bn;
                h0 = fminf(fmaxf(h0, 0.0f), 1.0f);
                h1 = fminf(fmaxf(h1, 0.0f), 1.0f);
                dp0[r] += h0 * wn;
                dp1[r] += h1 * wn;
            }
        }
        // reduce across the 16 N-columns (lanes within each half-wave)
        #pragma unroll
        for (int off = 1; off < 16; off <<= 1) {
            #pragma unroll
            for (int r = 0; r < 8; ++r) {
                dp0[r] += __shfl_xor(dp0[r], off, 32);
                dp1[r] += __shfl_xor(dp1[r], off, 32);
            }
        }
        if (mlane == 0) {
            #pragma unroll
            for (int r = 0; r < 8; ++r) {
                atomicAdd(&dotAcc[hi * 8 + r],      dp0[r]);
                atomicAdd(&dotAcc[16 + hi * 8 + r], dp1[r]);
            }
        }
    }

    __syncthreads();
    if (tid < MROWS) {
        int row = wg * MROWS + tid;
        if (row < batch) {
            float v = dotAcc[tid] + out_b[0];
            out[row] = 1.0f / (1.0f + expf(-v));
        }
    }
}

extern "C" void kernel_launch(void* const* d_in, const int* in_sizes, int n_in,
                              void* d_out, int out_size, void* d_ws, size_t ws_size,
                              hipStream_t stream) {
    const int*   stm    = (const int*)  d_in[0];
    const int*   nstm   = (const int*)  d_in[1];
    const float* values = (const float*)d_in[2];
    const float* ft_w   = (const float*)d_in[3];
    const float* ft_b   = (const float*)d_in[4];
    const float* out_w  = (const float*)d_in[5];
    const float* out_b  = (const float*)d_in[6];

    const int nnz   = in_sizes[0] / 2;
    const int batch = out_size;              // [B,1] floats
    const int mp    = nnz / batch;           // features per row (32)

    _Float16* w16 = (_Float16*)d_ws;         // 512*768*2 = 768 KB

    const int wconv_elems = FT_OUT * NFEAT;
    nn_convert_w<<<(wconv_elems + 255) / 256, 256, 0, stream>>>(ft_w, w16);

    const int ngroups = (batch + MROWS - 1) / MROWS;
    nn_forward<<<ngroups, 256, 0, stream>>>(stm, nstm, values, ft_b, out_w, out_b,
                                            w16, (float*)d_out, nnz, mp, batch);
}
